// Network_74131135529231
// MI455X (gfx1250) — matrix operand order repaired and checked by reading, not compiled, not run on hardware
//
#include <hip/hip_runtime.h>

// ---------------------------------------------------------------------------
// Model constants (match reference)
// ---------------------------------------------------------------------------
#define NN      4096      // neurons
#define NB      32        // batch
#define NSTEPS  300
#define NSTATE  (NB * NN) // 131072
#define NWIN    20
#define OUTN    (NB * NWIN * NN)

#define E_SYN   0.951229424500714f   // exp(-0.1/2.0)
#define DT_SYN  0.05f                // 0.1/2.0
#define E_TAU   0.904837418035960f   // exp(-0.1/1.0)
#define DT_TAU_ 0.1f                 // 0.1/1.0

// WMMA vector types (CDNA5 / gfx1250, wave32)
typedef __attribute__((ext_vector_type(16))) __bf16       v16bf;
typedef __attribute__((ext_vector_type(8)))  float        v8f;
typedef __attribute__((ext_vector_type(4)))  unsigned int v4u;

__device__ __forceinline__ unsigned short f32_to_bf16(float f) {
    unsigned int u = __float_as_uint(f);
    u += 0x7FFFu + ((u >> 16) & 1u);   // round-to-nearest-even
    return (unsigned short)(u >> 16);
}

// ---------------------------------------------------------------------------
// Zero / scale output
// ---------------------------------------------------------------------------
__global__ void k_zero(float* __restrict__ p, int n) {
    int i = blockIdx.x * 256 + threadIdx.x;
    if (i < n) p[i] = 0.0f;
}

__global__ void k_scale(float* __restrict__ p, int n) {
    int i = blockIdx.x * 256 + threadIdx.x;
    if (i < n) p[i] *= 0.1f;   // 1 / N_WINDOW
}

// ---------------------------------------------------------------------------
// Repack W (fp32 [K=4096][N=4096] row-major) into bf16 B-fragment-major
// layout for v_wmma_f32_16x16x32_bf16. For tile (tn, kc):
//   lane l<16 : dword d packs B[kc*32+2d][tn*16+l] , B[kc*32+2d+1][...]
//   lane l>=16: dword d packs B[kc*32+16+2d][tn*16+l-16], B[+1][...]
// Stored 8 contiguous dwords per lane -> two b128 loads per lane per chunk.
// Grid: 256 tn * 128 kc * 32 lanes * 8 dwords = 8,388,608 threads.
// ---------------------------------------------------------------------------
__global__ void k_repack(const float* __restrict__ W, unsigned int* __restrict__ Bp) {
    unsigned int idx = blockIdx.x * 256 + threadIdx.x;
    int d    = idx & 7;
    int lane = (idx >> 3) & 31;
    int kc   = (idx >> 8) & 127;
    int tn   = idx >> 15;
    int K = kc * 32 + ((lane >> 4) << 4) + 2 * d;
    int N = tn * 16 + (lane & 15);
    float w0 = W[(size_t)K * NN + N];
    float w1 = W[(size_t)(K + 1) * NN + N];
    Bp[idx] = (unsigned int)f32_to_bf16(w0) | ((unsigned int)f32_to_bf16(w1) << 16);
}

// ---------------------------------------------------------------------------
// Init: rates0 = relu(ff[:,0] + rec0); rec = rec0. Grid: 131072 threads.
// ---------------------------------------------------------------------------
__global__ void k_init(const float* __restrict__ ff, const float* __restrict__ rec0,
                       float* __restrict__ rates_f, unsigned short* __restrict__ rates_bf,
                       float* __restrict__ rec_f) {
    int idx = blockIdx.x * 256 + threadIdx.x;
    int b = idx >> 12, n = idx & (NN - 1);
    float r = ff[(size_t)b * NSTEPS * NN + n] + rec0[idx];
    r = fmaxf(r, 0.0f);
    rates_f[idx]  = r;
    rates_bf[idx] = f32_to_bf16(r);
    rec_f[idx]    = rec0[idx];
}

// ---------------------------------------------------------------------------
// GEMM step: hidden[32x4096] = rates[32x4096] @ W[4096x4096]  (bf16 WMMA)
// 256 waves, one N-tile (16 cols) each; each wave does BOTH batch M-tiles
// reusing one B fragment per k-chunk -> W read exactly once per step (L2).
// Grid: 32 blocks x 256 threads (8 waves/block).
// ---------------------------------------------------------------------------
__global__ void __launch_bounds__(256) k_gemm(const unsigned short* __restrict__ A,
                                              const unsigned int* __restrict__ Bp,
                                              float* __restrict__ H) {
    int gtid = blockIdx.x * 256 + threadIdx.x;
    int tn   = gtid >> 5;               // 0..255
    int lane = threadIdx.x & 31;
    int hi   = lane >> 4;               // 0: K-low half, 1: K-high half
    int l15  = lane & 15;

    // A fragment bases (k order used here, matched by Bp: lane holds row l15, one
    // contiguous 16-element run at k-offset hi*16)
    const v4u* a0p = (const v4u*)(A + (size_t)l15 * NN + (hi << 4));        // rows 0..15
    const v4u* a1p = (const v4u*)(A + (size_t)(16 + l15) * NN + (hi << 4)); // rows 16..31
    // B fragment base: ((tn*128 + kc)*32 + lane)*8 dwords, as v4u
    const v4u* bp = (const v4u*)Bp + (size_t)tn * 128 * 64 + lane * 2;

    v8f acc0 = {};
    v8f acc1 = {};

#pragma unroll 4
    for (int kc = 0; kc < 128; ++kc) {
        union { v4u u[2]; v16bf v; } a0, a1, b;
        const v4u* bt = bp + (size_t)kc * 64;
        __builtin_prefetch((const void*)(bt + 128), 0, 1);  // 2 chunks ahead (L2 stream)
        b.u[0]  = bt[0];
        b.u[1]  = bt[1];
        a0.u[0] = a0p[kc * 4];
        a0.u[1] = a0p[kc * 4 + 1];
        a1.u[0] = a1p[kc * 4];
        a1.u[1] = a1p[kc * 4 + 1];
        acc0 = __builtin_amdgcn_wmma_f32_16x16x32_bf16(false, a0.v, false, b.v,
                                                       (short)0, acc0, false, false);
        acc1 = __builtin_amdgcn_wmma_f32_16x16x32_bf16(false, a1.v, false, b.v,
                                                       (short)0, acc1, false, false);
    }

    // C/D layout: VGPR v, lanes 0-15 -> M=v, lanes 16-31 -> M=8+v; N = l15
    float* h0 = H + (size_t)(hi * 8) * NN + tn * 16 + l15;
    float* h1 = H + (size_t)(16 + hi * 8) * NN + tn * 16 + l15;
#pragma unroll
    for (int v = 0; v < 8; ++v) {
        h0[(size_t)v * NN] = acc0[v];
        h1[(size_t)v * NN] = acc1[v];
    }
}

// ---------------------------------------------------------------------------
// Per-step elementwise update + window accumulation. Grid: 131072 threads.
// ---------------------------------------------------------------------------
__global__ void k_update(const float* __restrict__ ff, const float* __restrict__ H,
                         float* __restrict__ rec_f, float* __restrict__ rates_f,
                         unsigned short* __restrict__ rates_bf, float* __restrict__ out,
                         int t) {
    int idx = blockIdx.x * 256 + threadIdx.x;
    int b = idx >> 12, n = idx & (NN - 1);
    float rec = rec_f[idx] * E_SYN + H[idx] * DT_SYN;
    float net = ff[(size_t)(b * NSTEPS + t) * NN + n] + rec;
    float nl  = fmaxf(net, 0.0f);
    float r   = rates_f[idx] * E_TAU + nl * DT_TAU_;
    rec_f[idx]    = rec;
    rates_f[idx]  = r;
    rates_bf[idx] = f32_to_bf16(r);
    // cumsum windows: w0 sums steps 90..100 (11 terms), then 10-step windows
    if (t >= 90 && t <= 290) {
        int w = (t <= 100) ? 0 : (t - 91) / 10;
        out[((size_t)b * NWIN + w) * NN + n] += r;
    }
}

// ---------------------------------------------------------------------------
// Host side
// ---------------------------------------------------------------------------
extern "C" void kernel_launch(void* const* d_in, const int* in_sizes, int n_in,
                              void* d_out, int out_size, void* d_ws, size_t ws_size,
                              hipStream_t stream) {
    const float* W    = (const float*)d_in[0];   // [4096,4096] fp32
    const float* ff   = (const float*)d_in[1];   // [32,300,4096] fp32
    const float* rec0 = (const float*)d_in[2];   // [32,4096] fp32
    float* out = (float*)d_out;                  // [32,20,4096] fp32

    char* ws = (char*)d_ws;
    const size_t BP_BYTES = (size_t)256 * 128 * 256 * 4;  // 32 MiB packed bf16 W
    unsigned int*   Bp       = (unsigned int*)ws;
    float*          rates_f  = (float*)(ws + BP_BYTES);
    float*          rec_f    = (float*)(ws + BP_BYTES + 1 * (size_t)NSTATE * 4);
    float*          hidden   = (float*)(ws + BP_BYTES + 2 * (size_t)NSTATE * 4);
    unsigned short* rates_bf = (unsigned short*)(ws + BP_BYTES + 3 * (size_t)NSTATE * 4);

    k_zero<<<(out_size + 255) / 256, 256, 0, stream>>>(out, out_size);
    k_repack<<<(256 * 128 * 256) / 256, 256, 0, stream>>>(W, Bp);
    k_init<<<NSTATE / 256, 256, 0, stream>>>(ff, rec0, rates_f, rates_bf, rec_f);

    for (int t = 0; t < NSTEPS; ++t) {
        k_gemm<<<32, 256, 0, stream>>>(rates_bf, Bp, hidden);
        k_update<<<NSTATE / 256, 256, 0, stream>>>(ff, hidden, rec_f, rates_f,
                                                   rates_bf, out, t);
    }

    k_scale<<<(out_size + 255) / 256, 256, 0, stream>>>(out, out_size);
}
